// Transformer_7370163880736
// MI455X (gfx1250) — matrix operand order repaired and checked by reading, not compiled, not run on hardware
//
#include <hip/hip_runtime.h>
#include <hip/hip_bf16.h>

// ---------------------------------------------------------------------------
// Transformer encoder layer for MI455X (gfx1250), bf16 WMMA everywhere.
//   B=8 S=1024 D=512 H=8 DH=64 FF=2048
// Compute-bound (~70 GFLOP vs ~80MB unique traffic -> ~875 FLOP/B), so all
// GEMMs (QKV proj, QK^T, PV, Wo, FFN1, FFN2) use v_wmma_f32_16x16x32_bf16.
// GEMM: 128x128 block tile, 8 waves x (32x64) wave tile, BK=32, double-
// buffered LDS, weights pre-transposed so staging is pure b128 copies.
// Workspace use: ~141 MB (fits the 192MB L2).
// ---------------------------------------------------------------------------

#define B_ 8
#define S_ 1024
#define D_ 512
#define H_ 8
#define DH_ 64
#define FF_ 2048
#define M_ (B_ * S_)   // 8192 token rows

typedef unsigned short us;  // bf16 bit pattern
typedef __attribute__((ext_vector_type(16))) __bf16 v16bf;
typedef __attribute__((ext_vector_type(8)))  float  v8f;

static __device__ __forceinline__ us f2bf(float f) {
    unsigned u = __builtin_bit_cast(unsigned, f);
    unsigned r = u + 0x7FFFu + ((u >> 16) & 1u);   // round-to-nearest-even
    return (us)(r >> 16);
}

static __device__ __forceinline__ v8f wmma_bf16(v16bf a, v16bf b, v8f c) {
    return __builtin_amdgcn_wmma_f32_16x16x32_bf16(
        /*neg_a=*/false, a, /*neg_b=*/false, b,
        /*c_mod=*/(short)0, c, /*reuse_a=*/false, /*reuse_b=*/false);
}

union FragU { v16bf v; uint4 u[2]; };

// A fragment (16x32 bf16) from a row-major [16 x strideH] array at column koff.
// Layout: lane<16 -> M=lane, halves 0..15 = K{0..15};
//         lanes 16..31: same M with K offset +16 (matches load_frag_bt).
static __device__ __forceinline__ v16bf load_frag_a(const us* base, int strideH,
                                                    int lane, int koff) {
    int m  = lane & 15;
    int kb = (lane & 16) ? 16 : 0;
    FragU f;
    f.u[0] = *(const uint4*)(base + m * strideH + koff + kb);
    f.u[1] = *(const uint4*)(base + m * strideH + koff + kb + 8);
    return f.v;
}

// B fragment (32x16 bf16) from an N-major ("Bt") [16 x strideH] array at koff.
// lane<16 -> N=lane, K=koff+0..15 ; lanes 16..31 -> K=koff+16..31.
static __device__ __forceinline__ v16bf load_frag_bt(const us* base, int strideH,
                                                     int lane, int koff) {
    int n  = lane & 15;
    int kb = (lane & 16) ? 16 : 0;
    const uint4* p = (const uint4*)(base + (size_t)n * strideH + koff + kb);
    FragU f;
    f.u[0] = p[0];
    f.u[1] = p[1];
    return f.v;
}

// ---------------------------------------------------------------------------
// Conversion / packing kernels (one-time cost; weights stored TRANSPOSED)
// ---------------------------------------------------------------------------
__global__ void k_f32_to_bf16(const float* __restrict__ in, us* __restrict__ out, int n) {
    for (int i = blockIdx.x * blockDim.x + threadIdx.x; i < n; i += gridDim.x * blockDim.x)
        out[i] = f2bf(in[i]);
}

// W[H,D,DH] -> out[N=H*DH, K=D] bf16 (transposed projection weights).
__global__ void k_pack_hde_t(const float* __restrict__ W, us* __restrict__ out, int n) {
    for (int i = blockIdx.x * blockDim.x + threadIdx.x; i < n; i += gridDim.x * blockDim.x) {
        int nn = i >> 9, d = i & 511;          // out[nn*512 + d]
        int h = nn >> 6, e = nn & 63;
        out[i] = f2bf(W[((size_t)h * D_ + d) * DH_ + e]);
    }
}

// W[K,N] -> out[N,K] bf16 (transposed dense weights).
__global__ void k_pack_t(const float* __restrict__ W, us* __restrict__ out,
                         int K, int N) {
    int total = K * N;
    for (int i = blockIdx.x * blockDim.x + threadIdx.x; i < total;
         i += gridDim.x * blockDim.x) {
        int nn = i / K, kk = i - nn * K;
        out[i] = f2bf(W[(size_t)kk * N + nn]);
    }
}

// ---------------------------------------------------------------------------
// bf16 WMMA GEMM: C[M,N] = A[M,K] * Bt[N,K]^T + bias, epilogue by MODE.
// 256 threads (8 waves). Block tile 128x128, BK=32, double-buffered LDS.
// Wave tile 32x64: 8 accumulators, 8 wmma per K-chunk.
// ---------------------------------------------------------------------------
enum { OUT_F32 = 0, OUT_BF16_RELU = 1, OUT_QK = 2, OUT_VT = 3 };

template <int MODE>
__global__ __launch_bounds__(256) void gemm_bf16(const us* __restrict__ A,
                                                 const us* __restrict__ Bt,
                                                 const float* __restrict__ bias,
                                                 void* __restrict__ out,
                                                 int Mdim, int Ndim, int Kdim) {
    __shared__ alignas(16) us As[2][128 * 32];   // A tile  [m][k]
    __shared__ alignas(16) us Bs[2][128 * 32];   // Bt tile [n][k]

    const int tid  = threadIdx.x;
    const int lane = tid & 31;
    const int w    = tid >> 5;
    const int wm   = w & 3;        // 4 M sub-blocks of 32
    const int wn   = w >> 2;       // 2 N sub-blocks of 64
    const int mBase = blockIdx.y * 128;
    const int nBase = blockIdx.x * 128;

    // Both stages are contiguous b128 copies (weights pre-transposed).
    auto stage = [&](int buf, int k0) {
#pragma unroll
        for (int i = 0; i < 2; ++i) {
            int linear = tid + i * 256;
            int row = linear >> 2, cb = (linear & 3) * 8;
            *(uint4*)&As[buf][row * 32 + cb] =
                *(const uint4*)&A[(size_t)(mBase + row) * Kdim + k0 + cb];
            *(uint4*)&Bs[buf][row * 32 + cb] =
                *(const uint4*)&Bt[(size_t)(nBase + row) * Kdim + k0 + cb];
        }
    };

    v8f acc[2][4];
#pragma unroll
    for (int mi = 0; mi < 2; ++mi)
#pragma unroll
        for (int j = 0; j < 4; ++j)
            acc[mi][j] = (v8f){0.f, 0.f, 0.f, 0.f, 0.f, 0.f, 0.f, 0.f};

    stage(0, 0);
    __syncthreads();

    int buf = 0;
    for (int k0 = 0; k0 < Kdim; k0 += 32) {
        if (k0 + 32 < Kdim) {
            stage(buf ^ 1, k0 + 32);   // overlap next-stage loads with wmma
            __builtin_prefetch(&A[(size_t)(mBase + (tid >> 2)) * Kdim + k0 + 64], 0, 0);
            __builtin_prefetch(&Bt[(size_t)(nBase + (tid >> 2)) * Kdim + k0 + 64], 0, 0);
        }
        v16bf a0 = load_frag_a(&As[buf][(wm * 32 + 0)  * 32], 32, lane, 0);
        v16bf a1 = load_frag_a(&As[buf][(wm * 32 + 16) * 32], 32, lane, 0);
        v16bf bf[4];
#pragma unroll
        for (int j = 0; j < 4; ++j)
            bf[j] = load_frag_bt(&Bs[buf][(wn * 64 + j * 16) * 32], 32, lane, 0);
#pragma unroll
        for (int j = 0; j < 4; ++j) {
            acc[0][j] = wmma_bf16(a0, bf[j], acc[0][j]);
            acc[1][j] = wmma_bf16(a1, bf[j], acc[1][j]);
        }
        __syncthreads();
        buf ^= 1;
    }

    // Epilogue. C layout: lane holds N=lane&15, rows r+(lane<16?0:8).
    const int colBase = nBase + wn * 64 + (lane & 15);
    const int rowSub  = (lane & 16) ? 8 : 0;
#pragma unroll
    for (int mi = 0; mi < 2; ++mi) {
        int rowBase = mBase + wm * 32 + mi * 16 + rowSub;
#pragma unroll
        for (int j = 0; j < 4; ++j) {
            int col = colBase + j * 16;
            float bval = bias ? bias[col] : 0.f;
#pragma unroll
            for (int r = 0; r < 8; ++r) {
                int row = rowBase + r;
                float v = acc[mi][j][r] + bval;
                if (MODE == OUT_F32) {
                    ((float*)out)[(size_t)row * Ndim + col] = v;
                } else if (MODE == OUT_BF16_RELU) {
                    ((us*)out)[(size_t)row * Ndim + col] = f2bf(fmaxf(v, 0.f));
                } else {
                    int b = row >> 10, s = row & 1023;      // S_=1024
                    int h = col >> 6,  e = col & 63;        // DH_=64
                    if (MODE == OUT_QK)  // [B,H,S,DH]
                        ((us*)out)[(((size_t)b * H_ + h) * S_ + s) * DH_ + e] = f2bf(v);
                    else                 // OUT_VT: [B,H,DH,S]
                        ((us*)out)[(((size_t)b * H_ + h) * DH_ + e) * S_ + s] = f2bf(v);
                }
            }
        }
    }
}

// ---------------------------------------------------------------------------
// Flash attention: 1 wave / block, 16 queries per block, online softmax.
// q:[B,H,S,DH] bf16, k:[B,H,S,DH] bf16, vT:[B,H,DH,S] bf16 -> ctx:[B,S,H*DH] bf16
// ---------------------------------------------------------------------------
__global__ __launch_bounds__(32) void attn_kernel(const us* __restrict__ q,
                                                  const us* __restrict__ k,
                                                  const us* __restrict__ vT,
                                                  us* __restrict__ ctx) {
    __shared__ alignas(16) us q_lds[16 * 64];   // Q tile
    __shared__ alignas(16) us p_lds[16 * 32];   // P tile (C->A layout transpose)

    const int lane = threadIdx.x;
    const int qt = blockIdx.x, h = blockIdx.y, b = blockIdx.z;
    const int bh = b * H_ + h;

    // Q tile rows are contiguous: straight 2KB copy into LDS.
    const us* qsrc = q + ((size_t)bh * S_ + qt * 16) * DH_;
    for (int i = lane; i < 128; i += 32)
        ((uint4*)q_lds)[i] = ((const uint4*)qsrc)[i];
    __syncthreads();

    const us* kbh = k  + (size_t)bh * S_ * DH_;
    const us* vbh = vT + (size_t)bh * DH_ * S_;
    const float scale = 0.125f;  // 1/sqrt(DH)

    float mrow[8], lrow[8];
#pragma unroll
    for (int r = 0; r < 8; ++r) { mrow[r] = -1e30f; lrow[r] = 0.f; }
    v8f o[4];
#pragma unroll
    for (int j = 0; j < 4; ++j) o[j] = (v8f){0.f, 0.f, 0.f, 0.f, 0.f, 0.f, 0.f, 0.f};

    for (int kc = 0; kc < S_; kc += 32) {
        // ---- scores: two 16x16 tiles, K-dim DH=64 -> 2 wmma each ----
        v8f s0 = {0.f, 0.f, 0.f, 0.f, 0.f, 0.f, 0.f, 0.f};
        v8f s1 = s0;
#pragma unroll
        for (int c = 0; c < 2; ++c) {
            v16bf a  = load_frag_a(q_lds, 64, lane, 32 * c);
            v16bf bA = load_frag_bt(kbh + (size_t)(kc)      * DH_, DH_, lane, 32 * c);
            v16bf bB = load_frag_bt(kbh + (size_t)(kc + 16) * DH_, DH_, lane, 32 * c);
            s0 = wmma_bf16(a, bA, s0);
            s1 = wmma_bf16(a, bB, s1);
        }

        // ---- online softmax; rows live in 16-lane halves -> shfl_xor 1,2,4,8 ----
        float p0[8], p1[8], fac[8];
#pragma unroll
        for (int r = 0; r < 8; ++r) {
            float x0 = s0[r] * scale, x1 = s1[r] * scale;
            float mx = fmaxf(x0, x1);
#pragma unroll
            for (int d = 1; d < 16; d <<= 1) mx = fmaxf(mx, __shfl_xor(mx, d, 32));
            float mnew = fmaxf(mrow[r], mx);
            fac[r] = __expf(mrow[r] - mnew);
            p0[r]  = __expf(x0 - mnew);
            p1[r]  = __expf(x1 - mnew);
            float rs = p0[r] + p1[r];
#pragma unroll
            for (int d = 1; d < 16; d <<= 1) rs += __shfl_xor(rs, d, 32);
            lrow[r] = lrow[r] * fac[r] + rs;
            mrow[r] = mnew;
        }
#pragma unroll
        for (int j = 0; j < 4; ++j)
#pragma unroll
            for (int r = 0; r < 8; ++r) o[j][r] *= fac[r];

        // ---- transpose P (C layout) -> A layout through LDS ----
        __syncthreads();  // guard vs previous iteration's p_lds fragment reads
        {
            int col = lane & 15;
            int rb  = (lane & 16) ? 8 : 0;
#pragma unroll
            for (int r = 0; r < 8; ++r) {
                p_lds[(r + rb) * 32 + col]      = f2bf(p0[r]);
                p_lds[(r + rb) * 32 + 16 + col] = f2bf(p1[r]);
            }
        }
        __syncthreads();

        // ---- O += P(16x32) * V(32x64): vT rows are Bt layout, contiguous in s ----
        v16bf ap = load_frag_a(p_lds, 32, lane, 0);
#pragma unroll
        for (int j = 0; j < 4; ++j) {
            v16bf vb = load_frag_bt(vbh + (size_t)(j * 16) * S_, S_, lane, kc);
            o[j] = wmma_bf16(ap, vb, o[j]);
        }
    }

    // normalize + write ctx in [B,S,H*DH] layout (A matrix for Wo GEMM)
    float inv[8];
#pragma unroll
    for (int r = 0; r < 8; ++r) inv[r] = 1.f / lrow[r];
    const int col = lane & 15;
    const int rb  = (lane & 16) ? 8 : 0;
#pragma unroll
    for (int j = 0; j < 4; ++j)
#pragma unroll
        for (int r = 0; r < 8; ++r) {
            int m = rb + r;
            ctx[((size_t)b * S_ + qt * 16 + m) * (H_ * DH_) + h * DH_ + j * 16 + col] =
                f2bf(o[j][r] * inv[r]);
        }
}

// ---------------------------------------------------------------------------
// Residual + LayerNorm over rows of 512. One block per row.
// ---------------------------------------------------------------------------
template <int OUTBF>
__global__ __launch_bounds__(256) void ln_kernel(const float* __restrict__ x,
                                                 const float* __restrict__ addv,
                                                 const float* __restrict__ g,
                                                 const float* __restrict__ bb,
                                                 void* __restrict__ outp) {
    __shared__ float red[256];
    const int row = blockIdx.x;
    const int t = threadIdx.x;
    const float* xr = x    + (size_t)row * D_;
    const float* ar = addv + (size_t)row * D_;

    float v0 = xr[t] + ar[t];
    float v1 = xr[t + 256] + ar[t + 256];

    red[t] = v0 + v1;
    __syncthreads();
    for (int o = 128; o > 0; o >>= 1) { if (t < o) red[t] += red[t + o]; __syncthreads(); }
    float mu = red[0] * (1.f / D_);
    __syncthreads();

    float d0 = v0 - mu, d1 = v1 - mu;
    red[t] = d0 * d0 + d1 * d1;
    __syncthreads();
    for (int o = 128; o > 0; o >>= 1) { if (t < o) red[t] += red[t + o]; __syncthreads(); }
    float rs = rsqrtf(red[0] * (1.f / D_) + 1e-6f);

    float y0 = d0 * rs * g[t] + bb[t];
    float y1 = d1 * rs * g[t + 256] + bb[t + 256];
    if (OUTBF) {
        us* o2 = (us*)outp;
        o2[(size_t)row * D_ + t]       = f2bf(y0);
        o2[(size_t)row * D_ + t + 256] = f2bf(y1);
    } else {
        float* o2 = (float*)outp;
        o2[(size_t)row * D_ + t]       = y0;
        o2[(size_t)row * D_ + t + 256] = y1;
    }
}

// ---------------------------------------------------------------------------
extern "C" void kernel_launch(void* const* d_in, const int* in_sizes, int n_in,
                              void* d_out, int out_size, void* d_ws, size_t ws_size,
                              hipStream_t stream) {
    (void)in_sizes; (void)n_in; (void)out_size; (void)ws_size;

    const float* x_q   = (const float*)d_in[0];
    const float* x_k   = (const float*)d_in[1];
    const float* x_v   = (const float*)d_in[2];
    const float* Wq    = (const float*)d_in[3];
    const float* bq    = (const float*)d_in[4];
    const float* Wk    = (const float*)d_in[5];
    const float* bk    = (const float*)d_in[6];
    const float* Wv    = (const float*)d_in[7];
    const float* bv    = (const float*)d_in[8];
    const float* Wo    = (const float*)d_in[9];
    const float* bo    = (const float*)d_in[10];
    const float* ln1_g = (const float*)d_in[11];
    const float* ln1_b = (const float*)d_in[12];
    const float* W1    = (const float*)d_in[13];
    const float* b1    = (const float*)d_in[14];
    const float* W2    = (const float*)d_in[15];
    const float* b2    = (const float*)d_in[16];
    const float* ln2_g = (const float*)d_in[17];
    const float* ln2_b = (const float*)d_in[18];

    // ---- workspace carve-out (~141 MB total) ----
    char* ws = (char*)d_ws;
    size_t off = 0;
    auto alloc = [&](size_t bytes) -> void* {
        void* p = ws + off;
        off = (off + bytes + 255) & ~(size_t)255;
        return p;
    };
    const size_t NTOK = (size_t)M_ * D_;          // 8192*512
    us* xq_bf = (us*)alloc(NTOK * 2);
    us* xk_bf = (us*)alloc(NTOK * 2);
    us* xv_bf = (us*)alloc(NTOK * 2);
    us* Wq_t  = (us*)alloc((size_t)512 * D_ * 2);   // [N=512, K=512]
    us* Wk_t  = (us*)alloc((size_t)512 * D_ * 2);
    us* Wv_t  = (us*)alloc((size_t)512 * D_ * 2);
    us* Wo_t  = (us*)alloc((size_t)D_ * 512 * 2);   // [N=512, K=512]
    us* W1_t  = (us*)alloc((size_t)FF_ * D_ * 2);   // [N=2048, K=512]
    us* W2_t  = (us*)alloc((size_t)D_ * FF_ * 2);   // [N=512, K=2048]
    us* qbuf  = (us*)alloc(NTOK * 2);               // [B,H,S,DH]
    us* kbuf  = (us*)alloc(NTOK * 2);               // [B,H,S,DH]
    us* vTbuf = (us*)alloc(NTOK * 2);               // [B,H,DH,S]
    us* ctx   = (us*)alloc(NTOK * 2);               // [B,S,H*DH]
    float* attn_out = (float*)alloc(NTOK * 4);      // [M, D]
    us* y_bf  = (us*)alloc(NTOK * 2);               // LN1 output, bf16
    us* h1    = (us*)alloc((size_t)M_ * FF_ * 2);   // FFN hidden, bf16
    float* ff = (float*)alloc(NTOK * 4);            // FFN output, f32

    // ---- 1. convert / pack (weights transposed to [N,K]) ----
    k_f32_to_bf16<<<4096, 256, 0, stream>>>(x_q, xq_bf, (int)NTOK);
    k_f32_to_bf16<<<4096, 256, 0, stream>>>(x_k, xk_bf, (int)NTOK);
    k_f32_to_bf16<<<4096, 256, 0, stream>>>(x_v, xv_bf, (int)NTOK);
    k_pack_hde_t<<<1024, 256, 0, stream>>>(Wq, Wq_t, 512 * D_);
    k_pack_hde_t<<<1024, 256, 0, stream>>>(Wk, Wk_t, 512 * D_);
    k_pack_hde_t<<<1024, 256, 0, stream>>>(Wv, Wv_t, 512 * D_);
    k_pack_t<<<1024, 256, 0, stream>>>(Wo, Wo_t, 512, D_);
    k_pack_t<<<4096, 256, 0, stream>>>(W1, W1_t, D_, FF_);
    k_pack_t<<<4096, 256, 0, stream>>>(W2, W2_t, FF_, D_);

    // ---- 2. QKV projections (bias layout [H,DH] flat == per-column bias) ----
    dim3 gProj(512 / 128, M_ / 128);
    gemm_bf16<OUT_QK><<<gProj, 256, 0, stream>>>(xq_bf, Wq_t, bq, qbuf,  M_, 512, D_);
    gemm_bf16<OUT_QK><<<gProj, 256, 0, stream>>>(xk_bf, Wk_t, bk, kbuf,  M_, 512, D_);
    gemm_bf16<OUT_VT><<<gProj, 256, 0, stream>>>(xv_bf, Wv_t, bv, vTbuf, M_, 512, D_);

    // ---- 3. flash attention ----
    attn_kernel<<<dim3(S_ / 16, H_, B_), 32, 0, stream>>>(qbuf, kbuf, vTbuf, ctx);

    // ---- 4. output projection ----
    gemm_bf16<OUT_F32><<<dim3(D_ / 128, M_ / 128), 256, 0, stream>>>(
        ctx, Wo_t, bo, attn_out, M_, D_, 512);

    // ---- 5. y = LN(x_q + attn_out) -> bf16 for FFN ----
    ln_kernel<1><<<M_, 256, 0, stream>>>(x_q, attn_out, ln1_g, ln1_b, y_bf);

    // ---- 6. FFN ----
    gemm_bf16<OUT_BF16_RELU><<<dim3(FF_ / 128, M_ / 128), 256, 0, stream>>>(
        y_bf, W1_t, b1, h1, M_, FF_, D_);
    gemm_bf16<OUT_F32><<<dim3(D_ / 128, M_ / 128), 256, 0, stream>>>(
        h1, W2_t, b2, ff, M_, D_, FF_);

    // ---- 7. out = LN(x_q + ff) ----
    ln_kernel<0><<<M_, 256, 0, stream>>>(x_q, ff, ln2_g, ln2_b, d_out);
}
